// SAGE_DIFFPOOL_79259326480894
// MI455X (gfx1250) — matrix operand order repaired and checked by reading, not compiled, not run on hardware
//
#include <hip/hip_runtime.h>

typedef __attribute__((ext_vector_type(16))) _Float16 v16h;
typedef __attribute__((ext_vector_type(8)))  _Float16 v8h;
typedef __attribute__((ext_vector_type(8)))  float    v8f;

#define NGRAPH 64
#define NNODE  512
#define NTOT   (NGRAPH * NNODE)      // 32768
#define NEDGE  (NGRAPH * 16384)      // 1048576
#define KCLUS  100
#define EPSBN  1e-5f

// ---------------------------------------------------------------- utilities

__global__ void k_zero_f32(float* __restrict__ p, long n) {
    long i = (long)blockIdx.x * blockDim.x + threadIdx.x;
    if (i < n) p[i] = 0.0f;
}

__global__ void k_zero_i32(int* __restrict__ p, int n) {
    int i = blockIdx.x * blockDim.x + threadIdx.x;
    if (i < n) p[i] = 0;
}

__global__ void k_degree(const int* __restrict__ dst, int* __restrict__ cnt) {
    int e = blockIdx.x * blockDim.x + threadIdx.x;
    if (e < NEDGE) atomicAdd(&cnt[dst[e]], 1);
}

__global__ void k_invdeg(const int* __restrict__ cnt, float* __restrict__ invdeg) {
    int i = blockIdx.x * blockDim.x + threadIdx.x;
    if (i < NTOT) {
        int c = cnt[i];
        invdeg[i] = 1.0f / (float)(c > 1 ? c : 1);
    }
}

// scatter-add of source features into destination accumulator (segment_sum)
__global__ void k_scatter_add(const float* __restrict__ xin, int fin,
                              const int* __restrict__ src, const int* __restrict__ dst,
                              float* __restrict__ agg) {
    int e = blockIdx.x * blockDim.x + threadIdx.x;
    if (e >= NEDGE) return;
    int s = src[e], d = dst[e];
    const float* xs = xin + (long)s * fin;
    float* ad = agg + (long)d * fin;
    for (int f = 0; f < fin; ++f)
        unsafeAtomicAdd(&ad[f], xs[f]);   // global_atomic_add_f32
}

// ---------------------------------------------------------------- packing
// 3-segment element fetchers (used only in streaming pack kernels)

__device__ __forceinline__ float fetchA3(const float* A0, int f0, const float* scale0,
                                         const float* A1, int f1,
                                         const float* A2, int f2,
                                         int m, int k) {
    if (k < f0) {
        float v = A0[(long)m * f0 + k];
        return scale0 ? v * scale0[m] : v;
    }
    k -= f0;
    if (k < f1) return A1[(long)m * f1 + k];
    k -= f1;
    return A2[(long)m * f2 + k];
}

__device__ __forceinline__ float fetchB3(const float* B0, int f0,
                                         const float* B1, int f1,
                                         const float* B2,
                                         int N, int k, int n) {
    if (k < f0) return B0[(long)k * N + n];
    k -= f0;
    if (k < f1) return B1[(long)k * N + n];
    k -= f1;
    return B2[(long)k * N + n];
}

// Ah[M x Kpad] f16 row-major, K zero-padded
__global__ void k_pack_a3(const float* __restrict__ A0, int f0, const float* __restrict__ scale0,
                          const float* __restrict__ A1, int f1,
                          const float* __restrict__ A2, int f2,
                          _Float16* __restrict__ Ah, int M, int Kpad) {
    long i = (long)blockIdx.x * blockDim.x + threadIdx.x;
    if (i >= (long)M * Kpad) return;
    int k = (int)(i % Kpad);
    int m = (int)(i / Kpad);
    int Ktot = f0 + f1 + f2;
    float v = (k < Ktot) ? fetchA3(A0, f0, scale0, A1, f1, A2, f2, m, k) : 0.0f;
    Ah[i] = (_Float16)v;
}

// Bt[Npad x Kpad] f16: Bt[n*Kpad+k] = B(k,n); zero-padded rows/cols
__global__ void k_pack_bt3(const float* __restrict__ B0, int f0,
                           const float* __restrict__ B1, int f1,
                           const float* __restrict__ B2, int f2,
                           int N, _Float16* __restrict__ Bt, int Npad, int Kpad) {
    long i = (long)blockIdx.x * blockDim.x + threadIdx.x;
    if (i >= (long)Npad * Kpad) return;
    int k = (int)(i % Kpad);
    int n = (int)(i / Kpad);
    int Ktot = f0 + f1 + f2;
    float v = (n < N && k < Ktot) ? fetchB3(B0, f0, B1, f1, B2, N, k, n) : 0.0f;
    Bt[i] = (_Float16)v;
}

// pooling einsum packs: At[b][112][512] = softmax^T, Bt[b][32][512] = x13^T
__global__ void k_pack_st(const float* __restrict__ S, _Float16* __restrict__ At) {
    long i = (long)blockIdx.x * blockDim.x + threadIdx.x;   // 64*112*512
    if (i >= (long)NGRAPH * 112 * NNODE) return;
    int k = (int)(i % NNODE);
    long t = i / NNODE;
    int m = (int)(t % 112);
    int b = (int)(t / 112);
    float v = (m < KCLUS) ? S[((long)b * NNODE + k) * KCLUS + m] : 0.0f;
    At[i] = (_Float16)v;
}

__global__ void k_pack_xt(const float* __restrict__ X, _Float16* __restrict__ Bt) {
    long i = (long)blockIdx.x * blockDim.x + threadIdx.x;   // 64*32*512
    if (i >= (long)NGRAPH * 32 * NNODE) return;
    int k = (int)(i % NNODE);
    long t = i / NNODE;
    int n = (int)(t % 32);
    int b = (int)(t / 32);
    float v = (n < 30) ? X[((long)b * NNODE + k) * 30 + n] : 0.0f;
    Bt[i] = (_Float16)v;
}

// ---------------------------------------------- packed-operand WMMA GEMM
// C[b] = Ah[b] (M x Kpad) @ Bt[b]^T (Kpad x Npad) + bias ; one wave per 16x16 tile.
// Fragments load contiguously: A lane needs halves k0+{0..7,16..23} (lanes 0-15)
// or k0+{8..15,24..31} (lanes 16-31); B lane needs the same k.

__global__ void k_gemm_wmma_packed(const _Float16* __restrict__ Ah, long strideAb,
                                   const _Float16* __restrict__ Bt, long strideBb,
                                   const float* __restrict__ bias,
                                   float* __restrict__ C, long strideCb,
                                   int Mstore, int N, int Kpad) {
    const int lane = threadIdx.x;      // 0..31
    const int half = lane >> 4;
    const int l16  = lane & 15;
    const int m0 = blockIdx.x * 16;
    const int n0 = blockIdx.y * 16;
    const int b  = blockIdx.z;

    const _Float16* arow = Ah + (long)b * strideAb + (long)(m0 + l16) * Kpad + half * 8;
    const _Float16* brow = Bt + (long)b * strideBb + (long)(n0 + l16) * Kpad + half * 8;

    v8f acc = {};
    for (int k0 = 0; k0 < Kpad; k0 += 32) {
        const v8h* ap = (const v8h*)arow;
        v8h alo = ap[0];                 // k0 + base .. +7
        v8h ahi = ap[2];                 // k0 + base+16 .. +23
        v16h a = __builtin_shufflevector(alo, ahi,
                 0, 1, 2, 3, 4, 5, 6, 7, 8, 9, 10, 11, 12, 13, 14, 15);
        const v8h* bp = (const v8h*)brow;
        v8h blo = bp[0];
        v8h bhi = bp[2];
        v16h bfr = __builtin_shufflevector(blo, bhi,
                 0, 1, 2, 3, 4, 5, 6, 7, 8, 9, 10, 11, 12, 13, 14, 15);
        acc = __builtin_amdgcn_wmma_f32_16x16x32_f16(false, a, false, bfr,
                                                     (short)0, acc, false, false);
        arow += 32;
        brow += 32;
    }
    const int cn = n0 + l16;
    float* Cb = C + (long)b * strideCb;
#pragma unroll
    for (int i = 0; i < 8; ++i) {
        int cm = m0 + i + half * 8;
        if (cm < Mstore && cn < N) {
            float v = acc[i];
            if (bias) v += bias[cn];
            Cb[(long)cm * N + cn] = v;
        }
    }
}

// ------------------------------------------------------------ batch norm

__global__ void k_bn_stats(const float* __restrict__ X, int M, int F,
                           float* __restrict__ mu, float* __restrict__ rs) {
    int f = blockIdx.x;
    float s = 0.0f, s2 = 0.0f;
    for (int m = threadIdx.x; m < M; m += blockDim.x) {
        float v = X[(long)m * F + f];
        s += v; s2 += v * v;
    }
    __shared__ float sh[256], sh2[256];
    sh[threadIdx.x] = s; sh2[threadIdx.x] = s2;
    __syncthreads();
    for (int off = blockDim.x >> 1; off > 0; off >>= 1) {
        if ((int)threadIdx.x < off) {
            sh[threadIdx.x]  += sh[threadIdx.x + off];
            sh2[threadIdx.x] += sh2[threadIdx.x + off];
        }
        __syncthreads();
    }
    if (threadIdx.x == 0) {
        float mean = sh[0] / (float)M;
        float var  = sh2[0] / (float)M - mean * mean;
        mu[f] = mean;
        rs[f] = rsqrtf(var + EPSBN);
    }
}

__global__ void k_bn_apply(float* __restrict__ X, int M, int F,
                           const float* __restrict__ mu, const float* __restrict__ rs,
                           const float* __restrict__ g, const float* __restrict__ bb) {
    long i = (long)blockIdx.x * blockDim.x + threadIdx.x;
    if (i < (long)M * F) {
        int f = (int)(i % F);
        X[i] = g[f] * (X[i] - mu[f]) * rs[f] + bb[f];
    }
}

// ------------------------------------------------------------ softmax (rows of 100)

__global__ void k_softmax_rows(float* __restrict__ S, int M, int F) {
    int m = blockIdx.x * blockDim.x + threadIdx.x;
    if (m >= M) return;
    float* r = S + (long)m * F;
    float mx = r[0];
    for (int f = 1; f < F; ++f) mx = fmaxf(mx, r[f]);
    float sum = 0.0f;
    for (int f = 0; f < F; ++f) { float e = expf(r[f] - mx); r[f] = e; sum += e; }
    float inv = 1.0f / sum;
    for (int f = 0; f < F; ++f) r[f] *= inv;
}

// ------------------------------- dense-SAGE aggregation: per-graph cluster mean

__global__ void k_cluster_mean(const float* __restrict__ X, float* __restrict__ A) {
    int b = blockIdx.x;
    __shared__ float mean[32];
    int c = threadIdx.x;
    if (c < 30) {
        float s = 0.0f;
        for (int k = 0; k < KCLUS; ++k) s += X[((long)b * KCLUS + k) * 30 + c];
        mean[c] = s * (1.0f / KCLUS);
    }
    __syncthreads();
    for (int idx = threadIdx.x; idx < KCLUS * 30; idx += blockDim.x)
        A[(long)b * KCLUS * 30 + idx] = mean[idx % 30];
}

// ------------------------------------------------------------ max pools

__global__ void k_max_nodes(const float* __restrict__ X1, const float* __restrict__ X2,
                            const float* __restrict__ X3, float* __restrict__ out) {
    int b = blockIdx.x, f = threadIdx.x;
    if (f >= 90) return;
    int sel = f / 30, ff = f % 30;
    const float* Xs = (sel == 0) ? X1 : ((sel == 1) ? X2 : X3);
    float mx = -3.402823e38f;
    for (int n = 0; n < NNODE; ++n)
        mx = fmaxf(mx, Xs[((long)b * NNODE + n) * 30 + ff]);
    out[b * 90 + f] = mx;
}

__global__ void k_max_clusters(const float* __restrict__ X1, const float* __restrict__ X2,
                               const float* __restrict__ X3, float* __restrict__ out) {
    int b = blockIdx.x, f = threadIdx.x;
    if (f >= 90) return;
    int sel = f / 30, ff = f % 30;
    const float* Xs = (sel == 0) ? X1 : ((sel == 1) ? X2 : X3);
    float mx = -3.402823e38f;
    for (int k = 0; k < KCLUS; ++k)
        mx = fmaxf(mx, Xs[((long)b * KCLUS + k) * 30 + ff]);
    out[b * 90 + f] = mx;
}

// ------------------------------------------------------------ head FCs

__global__ void k_head(const float* __restrict__ x1out, const float* __restrict__ x2out,
                       const float* __restrict__ W1, const float* __restrict__ b1,
                       const float* __restrict__ W2, const float* __restrict__ b2,
                       float* __restrict__ out) {
    int b = blockIdx.x;
    __shared__ float co[180];
    __shared__ float h[50];
    for (int i = threadIdx.x; i < 180; i += blockDim.x)
        co[i] = (i < 90) ? x1out[b * 90 + i] : x2out[b * 90 + (i - 90)];
    __syncthreads();
    for (int j = threadIdx.x; j < 50; j += blockDim.x) {
        float s = b1[j];
        for (int i = 0; i < 180; ++i) s += co[i] * W1[i * 50 + j];
        h[j] = fmaxf(s, 0.0f);
    }
    __syncthreads();
    for (int o = threadIdx.x; o < 6; o += blockDim.x) {
        float s = b2[o];
        for (int i = 0; i < 50; ++i) s += h[i] * W2[i * 6 + o];
        out[b * 6 + o] = s;
    }
    if (b == 0 && threadIdx.x == 0) out[NGRAPH * 6] = 0.0f;   // reg scalar
}

// ------------------------------------------------------------ launcher

extern "C" void kernel_launch(void* const* d_in, const int* in_sizes, int n_in,
                              void* d_out, int out_size, void* d_ws, size_t ws_size,
                              hipStream_t stream) {
    (void)in_sizes; (void)n_in; (void)out_size; (void)ws_size;

    const float* x  = (const float*)d_in[0];
    const int*   ei = (const int*)d_in[1];
    const int* src = ei;
    const int* dst = ei + NEDGE;

    int pi = 2;
#define P(name) const float* name = (const float*)d_in[pi++];
    P(c11Wl) P(c11Wr) P(c11b) P(n11g) P(n11b)
    P(c12Wl) P(c12Wr) P(c12b) P(n12g) P(n12b)
    P(c13Wl) P(c13Wr) P(c13b) P(n13g) P(n13b)
    P(p11Wl) P(p11Wr) P(p11b) P(np11g) P(np11b)
    P(p12Wl) P(p12Wr) P(p12b) P(np12g) P(np12b)
    P(p13Wl) P(p13Wr) P(p13b) P(np13g) P(np13b)
    P(pfcW)  P(pfcb)
    P(c21Wl) P(c21Wr) P(c21b) P(n21g) P(n21b)
    P(c22Wl) P(c22Wr) P(c22b) P(n22g) P(n22b)
    P(c23Wl) P(c23Wr) P(c23b) P(n23g) P(n23b)
    P(fc1W)  P(fc1b)
    P(fc2W)  P(fc2b)
#undef P

    // ---- workspace layout (floats; keep 16B alignment via multiples of 4)
    float* ws = (float*)d_ws;
    size_t o = 0;
    int*   cnt    = (int*)(ws + o); o += NTOT;
    float* invdeg = ws + o;         o += NTOT;
    float* agg    = ws + o;         o += (size_t)NTOT * 30;
    float* x11    = ws + o;         o += (size_t)NTOT * 30;
    float* x12    = ws + o;         o += (size_t)NTOT * 30;
    float* x13    = ws + o;         o += (size_t)NTOT * 30;
    float* s11    = ws + o;         o += (size_t)NTOT * 30;
    float* s12    = ws + o;         o += (size_t)NTOT * 30;
    float* s13    = ws + o;         o += (size_t)NTOT * 100;
    float* s1     = ws + o;         o += (size_t)NTOT * 100;   // logits -> softmax in place
    float* mu     = ws + o;         o += 128;
    float* rs     = ws + o;         o += 128;
    float* p1x    = ws + o;         o += (size_t)NGRAPH * KCLUS * 30;
    float* aggK   = ws + o;         o += (size_t)NGRAPH * KCLUS * 30;
    float* x21    = ws + o;         o += (size_t)NGRAPH * KCLUS * 30;
    float* x22    = ws + o;         o += (size_t)NGRAPH * KCLUS * 30;
    float* x23    = ws + o;         o += (size_t)NGRAPH * KCLUS * 30;
    float* x1out  = ws + o;         o += NGRAPH * 90;
    float* x2out  = ws + o;         o += NGRAPH * 90;
    // f16 packed operand buffers (2 halves per float slot)
    _Float16* Ah = (_Float16*)(ws + o); o += ((size_t)NTOT * 160) / 2;     // max M*Kpad
    _Float16* Bh = (_Float16*)(ws + o); o += (112 * 160) / 2 + 16;         // max Npad*Kpad
    _Float16* Ap = (_Float16*)(ws + o); o += ((size_t)NGRAPH * 112 * NNODE) / 2;
    _Float16* Bp = (_Float16*)(ws + o); o += ((size_t)NGRAPH * 32 * NNODE) / 2;

    // ---- degree (shared by all 6 node-level layers)
    k_zero_i32<<<NTOT / 256, 256, 0, stream>>>(cnt, NTOT);
    k_degree<<<NEDGE / 256, 256, 0, stream>>>(dst, cnt);
    k_invdeg<<<NTOT / 256, 256, 0, stream>>>(cnt, invdeg);

    auto zero = [&](float* p, size_t n) {
        k_zero_f32<<<(unsigned)((n + 255) / 256), 256, 0, stream>>>(p, (long)n);
    };

    // generic packed-WMMA GEMM: C = [A0*s|A1|A2] @ [B0;B1;B2] + bias
    auto gemm = [&](const float* A0, int f0, const float* scale0,
                    const float* A1, int f1, const float* A2, int f2,
                    const float* B0, const float* B1, const float* B2,
                    const float* bias, float* C, int M, int N) {
        int Ktot = f0 + f1 + f2;
        int Kpad = (Ktot + 31) & ~31;
        int Npad = (N + 15) & ~15;
        k_pack_a3<<<(unsigned)(((size_t)M * Kpad + 255) / 256), 256, 0, stream>>>(
            A0, f0, scale0, A1, f1, A2, f2, Ah, M, Kpad);
        k_pack_bt3<<<(unsigned)(((size_t)Npad * Kpad + 255) / 256), 256, 0, stream>>>(
            B0, f0, B1, f1, B2, f2, N, Bh, Npad, Kpad);
        dim3 grid(M / 16, Npad / 16, 1);
        k_gemm_wmma_packed<<<grid, 32, 0, stream>>>(Ah, 0, Bh, 0, bias, C, 0, M, N, Kpad);
    };

    // node-level SAGE layer: scatter-mean + WMMA GEMM + BN
    auto sage_node = [&](const float* xin, int fin,
                         const float* Wl, const float* Wr, const float* bb,
                         const float* g, const float* bbn,
                         float* xout, int fout) {
        zero(agg, (size_t)NTOT * fin);
        k_scatter_add<<<NEDGE / 256, 256, 0, stream>>>(xin, fin, src, dst, agg);
        gemm(agg, fin, invdeg, xin, fin, nullptr, 0, Wl, Wr, nullptr, bb, xout, NTOT, fout);
        k_bn_stats<<<fout, 256, 0, stream>>>(xout, NTOT, fout, mu, rs);
        k_bn_apply<<<(unsigned)(((size_t)NTOT * fout + 255) / 256), 256, 0, stream>>>(
            xout, NTOT, fout, mu, rs, g, bbn);
    };

    sage_node(x,   3,  c11Wl, c11Wr, c11b, n11g, n11b, x11, 30);
    sage_node(x11, 30, c12Wl, c12Wr, c12b, n12g, n12b, x12, 30);
    sage_node(x12, 30, c13Wl, c13Wr, c13b, n13g, n13b, x13, 30);

    sage_node(x,   3,  p11Wl, p11Wr, p11b, np11g, np11b, s11, 30);
    sage_node(s11, 30, p12Wl, p12Wr, p12b, np12g, np12b, s12, 30);
    sage_node(s12, 30, p13Wl, p13Wr, p13b, np13g, np13b, s13, 100);

    // pool_fc: [s11|s12|s13](32768x160) @ W(160x100) + b
    gemm(s11, 30, nullptr, s12, 30, s13, 100,
         pfcW, pfcW + 30 * KCLUS, pfcW + 60 * KCLUS, pfcb, s1, NTOT, KCLUS);

    // softmax over clusters, then P[b] = S[b]^T @ X13[b]  (batched WMMA, K=512)
    k_softmax_rows<<<NTOT / 256, 256, 0, stream>>>(s1, NTOT, KCLUS);
    k_pack_st<<<(unsigned)(((size_t)NGRAPH * 112 * NNODE + 255) / 256), 256, 0, stream>>>(s1, Ap);
    k_pack_xt<<<(unsigned)(((size_t)NGRAPH * 32 * NNODE + 255) / 256), 256, 0, stream>>>(x13, Bp);
    {
        dim3 grid(112 / 16, 32 / 16, NGRAPH);
        k_gemm_wmma_packed<<<grid, 32, 0, stream>>>(
            Ap, (long)112 * NNODE, Bp, (long)32 * NNODE, nullptr,
            p1x, (long)KCLUS * 30, KCLUS, 30, NNODE);
    }

    // dense-SAGE layers on [64,100,30] (fully connected pooled graph -> mean agg)
    auto sage_dense = [&](const float* xin,
                          const float* Wl, const float* Wr, const float* bb,
                          const float* g, const float* bbn, float* xout) {
        k_cluster_mean<<<NGRAPH, 128, 0, stream>>>(xin, aggK);
        gemm(aggK, 30, nullptr, xin, 30, nullptr, 0, Wl, Wr, nullptr, bb,
             xout, NGRAPH * KCLUS, 30);
        k_bn_stats<<<30, 256, 0, stream>>>(xout, NGRAPH * KCLUS, 30, mu, rs);
        k_bn_apply<<<(NGRAPH * KCLUS * 30 + 255) / 256, 256, 0, stream>>>(
            xout, NGRAPH * KCLUS, 30, mu, rs, g, bbn);
    };

    sage_dense(p1x, c21Wl, c21Wr, c21b, n21g, n21b, x21);
    sage_dense(x21, c22Wl, c22Wr, c22b, n22g, n22b, x22);
    sage_dense(x22, c23Wl, c23Wr, c23b, n23g, n23b, x23);

    // readouts + head
    k_max_nodes<<<NGRAPH, 96, 0, stream>>>(x11, x12, x13, x1out);
    k_max_clusters<<<NGRAPH, 96, 0, stream>>>(x21, x22, x23, x2out);
    k_head<<<NGRAPH, 64, 0, stream>>>(x1out, x2out, fc1W, fc1b, fc2W, fc2b,
                                      (float*)d_out);
}
